// InfiniTransformer_3642132267577
// MI455X (gfx1250) — matrix operand order repaired and checked by reading, not compiled, not run on hardware
//
#include <hip/hip_runtime.h>

// ---------------------------------------------------------------------------
// MI455X / gfx1250 Infini-attention transformer layer, bf16 WMMA everywhere.
// B=4 S=4096 D=1024 H=8 DK=DV=128 SEG=512 NSEG=8 DH=4096, M = B*S = 16384
// ---------------------------------------------------------------------------

typedef __attribute__((ext_vector_type(16))) __bf16 v16bf;
typedef __attribute__((ext_vector_type(8)))  float  v8f;
typedef __attribute__((ext_vector_type(4)))  unsigned int ui32x4;
typedef __attribute__((ext_vector_type(4)))  int i32x4;
typedef __attribute__((address_space(1))) i32x4 gi32x4;   // global v4i
typedef __attribute__((address_space(3))) i32x4 li32x4;   // LDS v4i

#define DEV __device__ __forceinline__

static constexpr int  B_   = 4;
static constexpr int  S_   = 4096;
static constexpr int  D_   = 1024;
static constexpr int  H_   = 8;
static constexpr int  DK_  = 128;
static constexpr int  SEG_ = 512;
static constexpr int  NSEG = 8;
static constexpr int  DH_  = 4096;
static constexpr long MR   = (long)B_ * S_;   // 16384 rows

#if __has_builtin(__builtin_amdgcn_global_load_async_to_lds_b128)
#define ASYNC_LDS 1
#endif

DEV unsigned short f2bf(float f) {                 // round-to-nearest-even
  unsigned int u = __builtin_bit_cast(unsigned int, f);
  u += 0x7fffu + ((u >> 16) & 1u);
  return (unsigned short)(u >> 16);
}
DEV float bf2f(unsigned short s) {
  unsigned int u = ((unsigned int)s) << 16;
  return __builtin_bit_cast(float, u);
}
// Assemble a 16-element bf16 fragment from two 16B-aligned 8-element chunks.
DEV v16bf ld_frag(const unsigned short* p0, const unsigned short* p1) {
  union { ui32x4 u[2]; v16bf v; } un;
  un.u[0] = *(const ui32x4*)p0;
  un.u[1] = *(const ui32x4*)p1;
  return un.v;
}
DEV v8f wmma_bf16(v16bf a, v16bf b, v8f c) {
  return __builtin_amdgcn_wmma_f32_16x16x32_bf16(false, a, false, b,
                                                 (short)0, c, false, false);
}
// 16B global -> LDS copy; async DMA path when the toolchain exposes it.
DEV void cp16_g2l(const unsigned short* g, unsigned short* l) {
#ifdef ASYNC_LDS
  __builtin_amdgcn_global_load_async_to_lds_b128((gi32x4*)g, (li32x4*)l, 0, 0);
#else
  *(ui32x4*)l = *(const ui32x4*)g;
#endif
}
DEV void async_wait_all() {
#ifdef ASYNC_LDS
#if __has_builtin(__builtin_amdgcn_s_wait_asynccnt)
  __builtin_amdgcn_s_wait_asynccnt(0);
#else
  asm volatile("s_wait_asynccnt 0" ::: "memory");
#endif
#endif
}

// ---------------------------------------------------------------------------
// fp32 -> bf16 weight conversion (grid-stride)
// ---------------------------------------------------------------------------
__global__ void cvt_bf16_kernel(const float* __restrict__ in,
                                unsigned short* __restrict__ out, long n) {
  long i = (long)blockIdx.x * blockDim.x + threadIdx.x;
  long stride = (long)gridDim.x * blockDim.x;
  for (; i < n; i += stride) out[i] = f2bf(in[i]);
}

// ---------------------------------------------------------------------------
// LayerNorm over D=1024, one block per row, bf16 output
// ---------------------------------------------------------------------------
__global__ __launch_bounds__(256) void ln_kernel(
    const float* __restrict__ x, const float* __restrict__ g,
    const float* __restrict__ be, unsigned short* __restrict__ out) {
  __shared__ float r1[256], r2[256];
  const int tid = threadIdx.x;
  const size_t row = blockIdx.x;
  const float* xp = x + row * D_;
  float v[4], s = 0.f, s2 = 0.f;
#pragma unroll
  for (int j = 0; j < 4; ++j) {
    v[j] = xp[tid + j * 256];
    s += v[j]; s2 += v[j] * v[j];
  }
  r1[tid] = s; r2[tid] = s2;
  __syncthreads();
  for (int off = 128; off > 0; off >>= 1) {
    if (tid < off) { r1[tid] += r1[tid + off]; r2[tid] += r2[tid + off]; }
    __syncthreads();
  }
  const float mu = r1[0] * (1.f / D_);
  const float var = r2[0] * (1.f / D_) - mu * mu;
  const float rstd = rsqrtf(var + 1e-5f);
#pragma unroll
  for (int j = 0; j < 4; ++j) {
    int c = tid + j * 256;
    out[row * D_ + c] = f2bf((v[j] - mu) * rstd * g[c] + be[c]);
  }
}

// ---------------------------------------------------------------------------
// Generic bf16 GEMM: C[M,N] = A[M,K] @ Bm[K,N], 128x128x32 tiles, 8 waves,
// double-buffered LDS with async global->LDS DMA for the A tile.
// mode 0: store bf16               mode 1: f32 = resid + acc
// mode 2: bf16 = gelu(acc + bias)  mode 3: f32 = resid + acc + bias
// ---------------------------------------------------------------------------
__global__ __launch_bounds__(256) void gemm_bf16_kernel(
    const unsigned short* __restrict__ A, const unsigned short* __restrict__ Bm,
    int M, int N, int K, int mode,
    const float* __restrict__ bias, const float* __restrict__ resid,
    float* __restrict__ outF, unsigned short* __restrict__ outB) {
  __shared__ unsigned short As[2][128][40];  // 20 KB (pad -> 16B aligned rows)
  __shared__ unsigned short Bs[2][128][40];  // transposed: Bs[buf][n][k]
  const int tid = threadIdx.x;
  const int wave = tid >> 5, lane = tid & 31;
  const int col16 = lane & 15, half = lane >> 4;
  const int mb = M / 128;
  const int bm = blockIdx.x % mb, bn = blockIdx.x / mb;
  const int mw = wave * 16;

  // staging roles
  const int rA = tid >> 1, hA = tid & 1;     // A: row rA, 16-elem chunk hA
  const int kB = tid >> 3, nB = tid & 7;     // B: k-row kB, 16 cols at nB*16
  const unsigned short* Agp = A + (size_t)(bm * 128 + rA) * K + hA * 16;
  const unsigned short* Bgp = Bm + (size_t)bn * 128 + nB * 16;

  v8f acc[8];
  const v8f vz = {0.f, 0.f, 0.f, 0.f, 0.f, 0.f, 0.f, 0.f};
#pragma unroll
  for (int f = 0; f < 8; ++f) acc[f] = vz;

#define STAGE_A(kk, buf)                                                \
  do {                                                                  \
    const unsigned short* gp = Agp + (kk);                              \
    cp16_g2l(gp,     &As[buf][rA][hA * 16]);                            \
    cp16_g2l(gp + 8, &As[buf][rA][hA * 16 + 8]);                        \
  } while (0)

#define STAGE_B(kk, buf)                                                \
  do {                                                                  \
    const unsigned short* gp = Bgp + (size_t)((kk) + kB) * N;           \
    union { ui32x4 u[2]; unsigned short s[16]; } tb;                    \
    tb.u[0] = *(const ui32x4*)gp;                                       \
    tb.u[1] = *(const ui32x4*)(gp + 8);                                 \
    _Pragma("unroll")                                                   \
    for (int j = 0; j < 16; ++j) Bs[buf][nB * 16 + j][kB] = tb.s[j];    \
  } while (0)

  STAGE_A(0, 0);
  STAGE_B(0, 0);
  async_wait_all();
  __syncthreads();

  int cb = 0;
  for (int kk = 0; kk < K; kk += 32) {
    const bool more = (kk + 32) < K;
    if (more) {
      STAGE_A(kk + 32, cb ^ 1);
      STAGE_B(kk + 32, cb ^ 1);
      if (kk + 64 < K) {  // pull the tile after next into cache
        __builtin_prefetch(Agp + kk + 64, 0, 1);
        __builtin_prefetch(Bgp + (size_t)(kk + 64 + kB) * N, 0, 1);
      }
    }
    const int ar = mw + col16;
    v16bf af = ld_frag(&As[cb][ar][half * 16], &As[cb][ar][half * 16 + 8]);
#pragma unroll
    for (int f = 0; f < 8; ++f) {
      const int n = f * 16 + col16;
      v16bf bfrag = ld_frag(&Bs[cb][n][half * 16], &Bs[cb][n][half * 16 + 8]);
      acc[f] = wmma_bf16(af, bfrag, acc[f]);
    }
    if (more) {
      async_wait_all();
      __syncthreads();
      cb ^= 1;
    }
  }
#undef STAGE_A
#undef STAGE_B

#pragma unroll
  for (int f = 0; f < 8; ++f) {
    const int n = bn * 128 + f * 16 + col16;
#pragma unroll
    for (int i = 0; i < 8; ++i) {
      const size_t m = (size_t)bm * 128 + mw + i + half * 8;
      float v = acc[f][i];
      if (mode == 0) {
        outB[m * N + n] = f2bf(v);
      } else if (mode == 1) {
        outF[m * N + n] = resid[m * N + n] + v;
      } else if (mode == 2) {
        v += bias[n];
        float t = 0.7978845608028654f * (v + 0.044715f * v * v * v);
        v = 0.5f * v * (1.f + tanhf(t));
        outB[m * N + n] = f2bf(v);
      } else {
        v += bias[n];
        outF[m * N + n] = resid[m * N + n] + v;
      }
    }
  }
}

// ---------------------------------------------------------------------------
// Pass 1: per (b,h,seg): segKV[dk][dv] = (elu(K)+1)^T @ V ; segZ[dk] = sum sk
// grid = B*H*NSEG = 256, 256 threads
// ---------------------------------------------------------------------------
__global__ __launch_bounds__(256) void seg_kvsum_kernel(
    const unsigned short* __restrict__ kb, const unsigned short* __restrict__ vb,
    float* __restrict__ segKV, float* __restrict__ segZ) {
  __shared__ unsigned short skT[128][72];  // skT[dk][key_in_chunk]
  __shared__ unsigned short vT[128][72];   // vT[dv][key_in_chunk]
  __shared__ float zsh[128];
  const int wg = blockIdx.x;
  const int seg = wg & 7, h = (wg >> 3) & 7, b = wg >> 6;
  const size_t baseRow = (size_t)b * S_ + (size_t)seg * SEG_;
  const int tid = threadIdx.x;
  const int wave = tid >> 5, lane = tid & 31;
  const int col16 = lane & 15, half = lane >> 4;
  if (tid < 128) zsh[tid] = 0.f;

  v8f acc[8];
  const v8f vz = {0.f, 0.f, 0.f, 0.f, 0.f, 0.f, 0.f, 0.f};
#pragma unroll
  for (int f = 0; f < 8; ++f) acc[f] = vz;
  __syncthreads();

  for (int sc = 0; sc < 8; ++sc) {  // 64 keys per chunk
    const int r = tid >> 2, part = tid & 3;  // key r, dims part*32..+31
    const unsigned short* kp = kb + (baseRow + sc * 64 + r) * (size_t)(H_ * DK_) + h * DK_ + part * 32;
    const unsigned short* vp = vb + (baseRow + sc * 64 + r) * (size_t)(H_ * DK_) + h * DK_ + part * 32;
#pragma unroll 8
    for (int j = 0; j < 32; ++j) {
      float kf = bf2f(kp[j]);
      float sk = kf > 0.f ? kf + 1.f : __expf(kf);  // elu(x)+1
      skT[part * 32 + j][r] = f2bf(sk);
      vT[part * 32 + j][r] = vp[j];
    }
    __syncthreads();
    if (tid < 128) {
      float s = 0.f;
      for (int j = 0; j < 64; ++j) s += bf2f(skT[tid][j]);
      zsh[tid] += s;
    }
    const int dkrow = wave * 16 + col16;
#pragma unroll
    for (int ks = 0; ks < 2; ++ks) {
      v16bf af = ld_frag(&skT[dkrow][ks * 32 + half * 16],
                         &skT[dkrow][ks * 32 + half * 16 + 8]);
#pragma unroll
      for (int f = 0; f < 8; ++f) {
        const int dv = f * 16 + col16;
        v16bf bfrag = ld_frag(&vT[dv][ks * 32 + half * 16],
                              &vT[dv][ks * 32 + half * 16 + 8]);
        acc[f] = wmma_bf16(af, bfrag, acc[f]);
      }
    }
    __syncthreads();
  }

  float* out = segKV + (size_t)wg * DK_ * DK_;
#pragma unroll
  for (int f = 0; f < 8; ++f)
#pragma unroll
    for (int i = 0; i < 8; ++i)
      out[(wave * 16 + i + half * 8) * DK_ + f * 16 + col16] = acc[f][i];
  if (tid < 128) segZ[(size_t)wg * DK_ + tid] = zsh[tid];
}

// ---------------------------------------------------------------------------
// Pass 2: exclusive prefix over segments -> per-segment mem / z state
// grid = B*H = 32
// ---------------------------------------------------------------------------
__global__ __launch_bounds__(256) void mem_prefix_kernel(
    const float* __restrict__ segKV, const float* __restrict__ segZ,
    float* __restrict__ memp, float* __restrict__ zp) {
  const int bh = blockIdx.x;
  const int tid = threadIdx.x;
  for (int e = tid; e < DK_ * DK_; e += 256) {
    float a = 0.f;
    for (int s = 0; s < NSEG; ++s) {
      const size_t idx = ((size_t)bh * NSEG + s) * (DK_ * DK_) + e;
      memp[idx] = a;
      a += segKV[idx];
    }
  }
  for (int e = tid; e < DK_; e += 256) {
    float a = 1.0f / (float)DK_;
    for (int s = 0; s < NSEG; ++s) {
      const size_t idx = ((size_t)bh * NSEG + s) * DK_ + e;
      zp[idx] = a;
      a += segZ[idx];
    }
  }
}

// ---------------------------------------------------------------------------
// Pass 3: local causal flash-attention + compressive-memory retrieval + gate
// grid = B*H*NSEG = 256, 256 threads (8 waves x 4 query strips of 16)
// ---------------------------------------------------------------------------
__global__ __launch_bounds__(256) void attn_kernel(
    const unsigned short* __restrict__ qb, const unsigned short* __restrict__ kb,
    const unsigned short* __restrict__ vb, const float* __restrict__ memp,
    const float* __restrict__ zp, const float* __restrict__ betas,
    unsigned short* __restrict__ ob) {
  __shared__ unsigned short vT[128][520];    // V^T for whole segment  (130 KB)
  __shared__ unsigned short memT[128][136];  // mem^T bf16             (34 KB)
  __shared__ unsigned short Pb[8][16][40];   // per-wave P tile
  __shared__ float rowst[8][16];             // per-wave sq@z row dots
  __shared__ float zsh[128];

  const int wg = blockIdx.x;
  const int seg = wg & 7, h = (wg >> 3) & 7, b = wg >> 6;
  const size_t baseRow = (size_t)b * S_ + (size_t)seg * SEG_;
  const int tid = threadIdx.x;
  const int wave = tid >> 5, lane = tid & 31;
  const int col16 = lane & 15, half = lane >> 4;
  const int HD = H_ * DK_;

  // stage V^T (vectorized read, scatter to LDS)
  for (int e = tid; e < SEG_ * 16; e += 256) {
    const int key = e >> 4, dvb = (e & 15) * 8;
    union { ui32x4 u; unsigned short s[8]; } t;
    t.u = *(const ui32x4*)(vb + (baseRow + key) * (size_t)HD + h * DK_ + dvb);
#pragma unroll
    for (int j = 0; j < 8; ++j) vT[dvb + j][key] = t.s[j];
  }
  // stage mem^T (f32 -> bf16)
  const float* mp = memp + (size_t)wg * DK_ * DK_;
  for (int e = tid; e < DK_ * DK_; e += 256) {
    const int dk = e >> 7, dv = e & 127;
    memT[dv][dk] = f2bf(mp[e]);
  }
  if (tid < 128) zsh[tid] = zp[(size_t)wg * DK_ + tid];
  __syncthreads();

  const float scale = 0.08838834764831845f;  // 1/sqrt(128)
  const v8f vz = {0.f, 0.f, 0.f, 0.f, 0.f, 0.f, 0.f, 0.f};

  for (int st = wave; st < 32; st += 8) {  // 16-query strips
    const int q0 = st * 16;
    const unsigned short* qp = qb + (baseRow + q0 + col16) * (size_t)HD + h * DK_;

    // A fragments for q (scores) and sq=elu(q)+1 (retrieval), plus sq@z dots
    v16bf aq[4], asq[4];
    float zdot = 0.f;
#pragma unroll
    for (int ks = 0; ks < 4; ++ks) {
      const int dkA = ks * 32 + half * 16;
      const int dkB = ks * 32 + half * 16 + 8;
      union { ui32x4 u[2]; v16bf v; unsigned short s[16]; } u;
      u.u[0] = *(const ui32x4*)(qp + dkA);
      u.u[1] = *(const ui32x4*)(qp + dkB);
      aq[ks] = u.v;
      union { ui32x4 u[2]; v16bf v; unsigned short s[16]; } w;
#pragma unroll
      for (int j = 0; j < 16; ++j) {
        float f = bf2f(u.s[j]);
        float sq = f > 0.f ? f + 1.f : __expf(f);
        const int dk = (j < 8) ? (dkA + j) : (dkB + j - 8);
        zdot += sq * zsh[dk];
        w.s[j] = f2bf(sq);
      }
      asq[ks] = w.v;
    }
    zdot += __shfl_xor(zdot, 16);
    if (half == 0) rowst[wave][col16] = zdot;
    asm volatile("s_wait_dscnt 0" ::: "memory");

    // retrieval: sq @ mem  (16 x 128)
    v8f racc[8];
#pragma unroll
    for (int f = 0; f < 8; ++f) racc[f] = vz;
#pragma unroll
    for (int ks = 0; ks < 4; ++ks)
#pragma unroll
      for (int f = 0; f < 8; ++f) {
        const int dv = f * 16 + col16;
        v16bf bm = ld_frag(&memT[dv][ks * 32 + half * 16],
                           &memT[dv][ks * 32 + half * 16 + 8]);
        racc[f] = wmma_bf16(asq[ks], bm, racc[f]);
      }

    // local causal attention, flash-style over 32-key chunks
    v8f oacc[8];
#pragma unroll
    for (int f = 0; f < 8; ++f) oacc[f] = vz;
    float mrun[8], lrun[8];
#pragma unroll
    for (int i = 0; i < 8; ++i) { mrun[i] = -1e30f; lrun[i] = 0.f; }

    const int nch = st / 2 + 1;
    for (int kc = 0; kc < nch; ++kc) {
      const int kb0 = kc * 32;
      v8f s0 = vz, s1 = vz;
#pragma unroll
      for (int ks = 0; ks < 4; ++ks) {
        const unsigned short* kp0 =
            kb + (baseRow + kb0 + col16) * (size_t)HD + h * DK_ + ks * 32 + half * 16;
        const unsigned short* kp1 =
            kb + (baseRow + kb0 + 16 + col16) * (size_t)HD + h * DK_ + ks * 32 + half * 16;
        s0 = wmma_bf16(aq[ks], ld_frag(kp0, kp0 + 8), s0);
        s1 = wmma_bf16(aq[ks], ld_frag(kp1, kp1 + 8), s1);
      }
      float fac[8];
#pragma unroll
      for (int i = 0; i < 8; ++i) {
        const int qr = q0 + i + half * 8;
        float v0 = s0[i] * scale; if (kb0 + col16 > qr) v0 = -1e30f;
        float v1 = s1[i] * scale; if (kb0 + 16 + col16 > qr) v1 = -1e30f;
        float mx = fmaxf(v0, v1);
        mx = fmaxf(mx, __shfl_xor(mx, 1));
        mx = fmaxf(mx, __shfl_xor(mx, 2));
        mx = fmaxf(mx, __shfl_xor(mx, 4));
        mx = fmaxf(mx, __shfl_xor(mx, 8));
        const float mnew = fmaxf(mrun[i], mx);
        const float p0 = __expf(v0 - mnew), p1 = __expf(v1 - mnew);
        float rs = p0 + p1;
        rs += __shfl_xor(rs, 1); rs += __shfl_xor(rs, 2);
        rs += __shfl_xor(rs, 4); rs += __shfl_xor(rs, 8);
        const float sc2 = __expf(mrun[i] - mnew);
        lrun[i] = lrun[i] * sc2 + rs;
        mrun[i] = mnew;
        fac[i] = sc2;
        const int row = i + half * 8;
        Pb[wave][row][col16]      = f2bf(p0);
        Pb[wave][row][16 + col16] = f2bf(p1);
      }
#pragma unroll
      for (int f = 0; f < 8; ++f)
#pragma unroll
        for (int i = 0; i < 8; ++i) oacc[f][i] *= fac[i];
      asm volatile("s_wait_dscnt 0" ::: "memory");
      v16bf ap = ld_frag(&Pb[wave][col16][half * 16],
                         &Pb[wave][col16][half * 16 + 8]);
#pragma unroll
      for (int f = 0; f < 8; ++f) {
        const int dv = f * 16 + col16;
        v16bf bv = ld_frag(&vT[dv][kb0 + half * 16],
                           &vT[dv][kb0 + half * 16 + 8]);
        oacc[f] = wmma_bf16(ap, bv, oacc[f]);
      }
    }

    // combine: out = g * (sq@mem)/(sq@z) + (1-g) * softmax@V
    asm volatile("s_wait_dscnt 0" ::: "memory");
    float zrow[8];
#pragma unroll
    for (int i = 0; i < 8; ++i) zrow[i] = rowst[wave][i + half * 8];
#pragma unroll
    for (int f = 0; f < 8; ++f) {
      const int dv = f * 16 + col16;
      const float g = 1.f / (1.f + __expf(-betas[h * DK_ + dv]));
#pragma unroll
      for (int i = 0; i < 8; ++i) {
        const float am = racc[f][i] / zrow[i];
        const float al = oacc[f][i] / lrun[i];
        ob[(baseRow + q0 + i + half * 8) * (size_t)HD + h * DK_ + dv] =
            f2bf(g * am + (1.f - g) * al);
      }
    }
  }
}

// ---------------------------------------------------------------------------
// Host-side orchestration
// ---------------------------------------------------------------------------
extern "C" void kernel_launch(void* const* d_in, const int* in_sizes, int n_in,
                              void* d_out, int out_size, void* d_ws, size_t ws_size,
                              hipStream_t stream) {
  (void)in_sizes; (void)n_in; (void)out_size; (void)ws_size;
  const float* x     = (const float*)d_in[0];
  const float* gam   = (const float*)d_in[1];
  const float* bet   = (const float*)d_in[2];
  const float* Wq    = (const float*)d_in[3];
  const float* Wk    = (const float*)d_in[4];
  const float* Wv    = (const float*)d_in[5];
  const float* Wo    = (const float*)d_in[6];
  const float* betas = (const float*)d_in[7];
  const float* ffw1  = (const float*)d_in[8];
  const float* ffb1  = (const float*)d_in[9];
  const float* ffw2  = (const float*)d_in[10];
  const float* ffb2  = (const float*)d_in[11];
  float* out = (float*)d_out;

  char* ws = (char*)d_ws;
  const size_t SZ_BF_MD = (size_t)MR * D_ * 2;         // 32 MB
  size_t off_h   = 0;                                   // h / h2 (bf16)
  size_t off_q   = off_h + SZ_BF_MD;
  size_t off_k   = off_q + SZ_BF_MD;
  size_t off_v   = off_k + SZ_BF_MD;
  size_t off_att = off_v + SZ_BF_MD;
  size_t off_ff1 = off_q;                               // overlays q,k,v,att
  size_t off_x2  = off_att + SZ_BF_MD;                  // f32 residual stream
  size_t off_wq  = off_x2 + (size_t)MR * D_ * 4;
  size_t off_wk  = off_wq + (size_t)D_ * D_ * 2;
  size_t off_wv  = off_wk + (size_t)D_ * D_ * 2;
  size_t off_wo  = off_wv + (size_t)D_ * D_ * 2;
  size_t off_w1  = off_wo + (size_t)D_ * D_ * 2;
  size_t off_w2  = off_w1 + (size_t)D_ * DH_ * 2;
  size_t off_skv = off_w2 + (size_t)D_ * DH_ * 2;
  size_t off_sz  = off_skv + (size_t)B_ * H_ * NSEG * DK_ * DK_ * 4;
  size_t off_mp  = off_sz + (size_t)B_ * H_ * NSEG * DK_ * 4;
  size_t off_zp  = off_mp + (size_t)B_ * H_ * NSEG * DK_ * DK_ * 4;

  unsigned short* h_bf   = (unsigned short*)(ws + off_h);
  unsigned short* q_bf   = (unsigned short*)(ws + off_q);
  unsigned short* k_bf   = (unsigned short*)(ws + off_k);
  unsigned short* v_bf   = (unsigned short*)(ws + off_v);
  unsigned short* att_bf = (unsigned short*)(ws + off_att);
  unsigned short* ff1_bf = (unsigned short*)(ws + off_ff1);
  float*          x2     = (float*)(ws + off_x2);
  unsigned short* wq_bf  = (unsigned short*)(ws + off_wq);
  unsigned short* wk_bf  = (unsigned short*)(ws + off_wk);
  unsigned short* wv_bf  = (unsigned short*)(ws + off_wv);
  unsigned short* wo_bf  = (unsigned short*)(ws + off_wo);
  unsigned short* w1_bf  = (unsigned short*)(ws + off_w1);
  unsigned short* w2_bf  = (unsigned short*)(ws + off_w2);
  float*          segKV  = (float*)(ws + off_skv);
  float*          segZ   = (float*)(ws + off_sz);
  float*          memp   = (float*)(ws + off_mp);
  float*          zpre   = (float*)(ws + off_zp);

  // 1) weights -> bf16
  cvt_bf16_kernel<<<2048, 256, 0, stream>>>(Wq,   wq_bf, (long)D_ * D_);
  cvt_bf16_kernel<<<2048, 256, 0, stream>>>(Wk,   wk_bf, (long)D_ * D_);
  cvt_bf16_kernel<<<2048, 256, 0, stream>>>(Wv,   wv_bf, (long)D_ * D_);
  cvt_bf16_kernel<<<2048, 256, 0, stream>>>(Wo,   wo_bf, (long)D_ * D_);
  cvt_bf16_kernel<<<4096, 256, 0, stream>>>(ffw1, w1_bf, (long)D_ * DH_);
  cvt_bf16_kernel<<<4096, 256, 0, stream>>>(ffw2, w2_bf, (long)D_ * DH_);

  // 2) LN1
  ln_kernel<<<(int)MR, 256, 0, stream>>>(x, gam, bet, h_bf);

  // 3) Q/K/V projections
  const int gQKV = (MR / 128) * (D_ / 128);  // 1024
  gemm_bf16_kernel<<<gQKV, 256, 0, stream>>>(h_bf, wq_bf, (int)MR, D_, D_, 0,
                                             nullptr, nullptr, nullptr, q_bf);
  gemm_bf16_kernel<<<gQKV, 256, 0, stream>>>(h_bf, wk_bf, (int)MR, D_, D_, 0,
                                             nullptr, nullptr, nullptr, k_bf);
  gemm_bf16_kernel<<<gQKV, 256, 0, stream>>>(h_bf, wv_bf, (int)MR, D_, D_, 0,
                                             nullptr, nullptr, nullptr, v_bf);

  // 4) compressive-memory segment sums + prefix
  seg_kvsum_kernel<<<B_ * H_ * NSEG, 256, 0, stream>>>(k_bf, v_bf, segKV, segZ);
  mem_prefix_kernel<<<B_ * H_, 256, 0, stream>>>(segKV, segZ, memp, zpre);

  // 5) attention (local flash + memory retrieval + gate)
  attn_kernel<<<B_ * H_ * NSEG, 256, 0, stream>>>(q_bf, k_bf, v_bf, memp, zpre,
                                                  betas, att_bf);

  // 6) output projection + residual
  gemm_bf16_kernel<<<gQKV, 256, 0, stream>>>(att_bf, wo_bf, (int)MR, D_, D_, 1,
                                             nullptr, x, x2, nullptr);

  // 7) LN2 + FFN
  ln_kernel<<<(int)MR, 256, 0, stream>>>(x2, gam, bet, h_bf);
  const int gFF1 = (MR / 128) * (DH_ / 128);  // 4096
  gemm_bf16_kernel<<<gFF1, 256, 0, stream>>>(h_bf, w1_bf, (int)MR, DH_, D_, 2,
                                             ffb1, nullptr, nullptr, ff1_bf);
  gemm_bf16_kernel<<<gQKV, 256, 0, stream>>>(ff1_bf, w2_bf, (int)MR, D_, DH_, 3,
                                             ffb2, x2, out, nullptr);
}
